// ChordHMM_11673721111000
// MI455X (gfx1250) — matrix operand order repaired and checked by reading, not compiled, not run on hardware
//
#include <hip/hip_runtime.h>

// ---------------------------------------------------------------------------
// HMM forward NLL on gfx1250 (MI455X).
// Whole scan runs in the log2 domain so v_exp_f32/v_log_f32 are used raw:
//   m2      = rowmax(alpha2)                       (per batch, log2 units)
//   p       = exp2(alpha2 - m2)          (f16)
//   s       = p @ exp(log_trans)         (v_wmma_f32_16x16x32_f16, f32 acc)
//   alpha2' = log2(s) + 0.8*log2e*em - lsePre;  carry2 += m2
// Final: nll[b] = -ln2 * (log2(sum exp2(alpha2 - m2)) + carry2)
// ---------------------------------------------------------------------------

typedef __attribute__((ext_vector_type(16))) _Float16 v16h;
typedef __attribute__((ext_vector_type(8)))  _Float16 v8h;
typedef __attribute__((ext_vector_type(8)))  float    v8f;

#define S_DIM   170
#define SN      176   // N padded: 11 tiles of 16
#define SK      192   // K padded: 6 chunks of 32
#define BATCH   32
#define TLEN    4000
#define NWAVES  11
#define NTHREADS (NWAVES*32)

#define LOG2E     1.4426950408889634f
#define LN2       0.6931471805599453f
#define EMW_LOG2E (0.8f * LOG2E)        // emission weight folded into log2 domain
#define T_LOG2E   (2.0f * LOG2E)        // 1/temperature folded into log2 domain

#define APAD 196      // alphaLds row stride (f32), breaks bank conflicts
#define PPAD 200      // pExp row stride (f16), breaks bank conflicts

#define TSTRIDE ((size_t)TLEN * S_DIM)  // per-batch emission row stride (floats)

// workspace layout (bytes)
#define WS_TRANS    0        // [SN][SK] f16  = 67584 B
#define WS_LOGSTART 67584    // [S_DIM] f32 (log2 units)
#define WS_LSE      68608    // [BATCH*TLEN] f32 = 512000 B (pre-scaled: 0.8*log2e*lse)

__device__ __forceinline__ float fast_exp2(float x) { return __builtin_amdgcn_exp2f(x); }
__device__ __forceinline__ float fast_log2(float x) { return __builtin_amdgcn_logf(x); }

// ---------------------------------------------------------------------------
__global__ void zero_u32_kernel(uint32_t* p, int n) {
  int i = blockIdx.x * blockDim.x + threadIdx.x;
  if (i < n) p[i] = 0u;
}

// one wave per transition row i: softmax(raw/T) in log2 domain, store exp transposed+padded
__global__ void trans_kernel(const float* __restrict__ raw, _Float16* __restrict__ trans_w) {
  int i = blockIdx.x;
  int lane = threadIdx.x;
  float y[6]; float m = -INFINITY;
#pragma unroll
  for (int q = 0; q < 6; ++q) {
    int j = lane + 32 * q;
    y[q] = (j < S_DIM) ? raw[(size_t)i * S_DIM + j] * T_LOG2E : -INFINITY;
    m = fmaxf(m, y[q]);
  }
#pragma unroll
  for (int o = 16; o; o >>= 1) m = fmaxf(m, __shfl_xor(m, o, 32));
  float s = 0.f;
#pragma unroll
  for (int q = 0; q < 6; ++q) s += fast_exp2(y[q] - m);
#pragma unroll
  for (int o = 16; o; o >>= 1) s += __shfl_xor(s, o, 32);
  float lse2 = m + fast_log2(s);
#pragma unroll
  for (int q = 0; q < 6; ++q) {
    int j = lane + 32 * q;
    if (j < S_DIM) trans_w[(size_t)j * SK + i] = (_Float16)fast_exp2(y[q] - lse2);
  }
}

// log_softmax(start) in log2 units
__global__ void logstart_kernel(const float* __restrict__ sp, float* __restrict__ ls) {
  int lane = threadIdx.x;
  float y[6]; float m = -INFINITY;
#pragma unroll
  for (int q = 0; q < 6; ++q) {
    int j = lane + 32 * q;
    y[q] = (j < S_DIM) ? sp[j] * LOG2E : -INFINITY;
    m = fmaxf(m, y[q]);
  }
#pragma unroll
  for (int o = 16; o; o >>= 1) m = fmaxf(m, __shfl_xor(m, o, 32));
  float s = 0.f;
#pragma unroll
  for (int q = 0; q < 6; ++q) s += fast_exp2(y[q] - m);
#pragma unroll
  for (int o = 16; o; o >>= 1) s += __shfl_xor(s, o, 32);
  float lse2 = m + fast_log2(s);
#pragma unroll
  for (int q = 0; q < 6; ++q) {
    int j = lane + 32 * q;
    if (j < S_DIM) ls[j] = y[q] - lse2;
  }
}

// per-(b,t) emission logsumexp, pre-scaled by 0.8*log2e: one wave per row, 8 rows/block
__global__ void emlse_kernel(const float* __restrict__ em, float* __restrict__ lse) {
  int row = blockIdx.x * 8 + (threadIdx.x >> 5);
  int lane = threadIdx.x & 31;
  const float* p = em + (size_t)row * S_DIM;
  float y[6]; float m = -INFINITY;
#pragma unroll
  for (int q = 0; q < 6; ++q) {
    int j = lane + 32 * q;
    y[q] = (j < S_DIM) ? p[j] * LOG2E : -INFINITY;
    m = fmaxf(m, y[q]);
  }
#pragma unroll
  for (int o = 16; o; o >>= 1) m = fmaxf(m, __shfl_xor(m, o, 32));
  float s = 0.f;
#pragma unroll
  for (int q = 0; q < 6; ++q) s += fast_exp2(y[q] - m);
#pragma unroll
  for (int o = 16; o; o >>= 1) s += __shfl_xor(s, o, 32);
  if (lane == 0) lse[row] = 0.8f * (m + fast_log2(s));   // = EMW_LOG2E * lse_natural
}

// ---------------------------------------------------------------------------
// The sequential scan: 2 blocks x 11 waves; block handles 16 batch rows.
__launch_bounds__(NTHREADS, 1)
__global__ void scan_kernel(const float* __restrict__ em,
                            const float* __restrict__ logstart,
                            const _Float16* __restrict__ trans_w,
                            const float* __restrict__ lse,
                            float* __restrict__ out) {
  __shared__ __align__(16) float     alphaLds[16 * APAD];
  __shared__ __align__(16) _Float16  pExp[16 * PPAD];
  __shared__ float wmax[NWAVES][16];
  __shared__ float mLds[16];
  __shared__ float carry[16];

  const int tid  = threadIdx.x;
  const int wave = tid >> 5;
  const int lane = tid & 31;
  const int hi   = (lane >= 16) ? 1 : 0;
  const int ln16 = lane & 15;
  const int wg   = blockIdx.x;           // batches [16*wg, 16*wg+16)
  const int n    = wave * 16 + ln16;     // this lane's destination state column
  const bool nvalid = (n < S_DIM);
  const int nc   = nvalid ? n : (S_DIM - 1);  // clamped: loads always safe, branchless

  // init LDS (padding columns >=176 stay -inf forever)
  for (int e = tid; e < 16 * APAD; e += NTHREADS) alphaLds[e] = -INFINITY;
  if (tid < 16) carry[tid] = 0.f;

  // resident B fragments: B[k,n] = exp(log_trans[k][n]); halves p <-> K = 32c + 16*hi + p
  v16h bfrag[6];
#pragma unroll
  for (int c = 0; c < 6; ++c) {
    const v8h* bp = (const v8h*)(trans_w + (size_t)n * SK + 32 * c + hi * 16);
    v16h f;
    ((v8h*)&f)[0] = bp[0];
    ((v8h*)&f)[1] = bp[1];
    bfrag[c] = f;
  }

  float aval[8];   // C-tile column values: row M = r + 8*hi, col = n  (log2 units)

  // shared tail: publish alpha, reduce row-max, stage exp2() in A-layout-friendly LDS
  auto tail = [&]() {
    const int row0 = hi * 8;
#pragma unroll
    for (int r = 0; r < 8; ++r)
      alphaLds[(row0 + r) * APAD + n] = aval[r];
    // per-row max across the 16 lanes of this half-wave
#pragma unroll
    for (int r = 0; r < 8; ++r) {
      float v = aval[r];
      v = fmaxf(v, __shfl_xor(v, 1, 32));
      v = fmaxf(v, __shfl_xor(v, 2, 32));
      v = fmaxf(v, __shfl_xor(v, 4, 32));
      v = fmaxf(v, __shfl_xor(v, 8, 32));
      if (ln16 == 0) wmax[wave][row0 + r] = v;
    }
    __syncthreads();                                   // B1
    // pExp stage: thread -> (row = tid&15, k-pairs kp = tid>>4 + 22*q); 352 = 16*22
    {
      const int row = tid & 15;
      const int kp0 = tid >> 4;   // 0..21
      float m = wmax[0][row];
#pragma unroll
      for (int w = 1; w < NWAVES; ++w) m = fmaxf(m, wmax[w][row]);
      if (tid < 16) { mLds[row] = m; carry[row] += m; }
      const float* arow = alphaLds + row * APAD;
      _Float16*    prow = pExp + row * PPAD;
      for (int kp = kp0; kp < SK / 2; kp += 22) {
        const int k = 2 * kp;
        float a = fast_exp2(arow[k]     - m);
        float b = fast_exp2(arow[k + 1] - m);
        auto pk = __builtin_amdgcn_cvt_pkrtz(a, b);    // v_cvt_pk_rtz_f16_f32
        *reinterpret_cast<decltype(pk)*>(prow + k) = pk;
      }
    }
    __syncthreads();                                   // B2
  };

  // running base pointers (advance by one emission row / one lse entry per step);
  // per-r offsets r*TSTRIDE (r<4) fit the 24-bit immediate offset field.
  const int b0 = wg * 16 + hi * 8;
  const float* pe0 = em + (size_t)b0 * TSTRIDE + nc;   // rows r = 0..3
  const float* pe4 = pe0 + 4 * TSTRIDE;                // rows r = 4..7
  const float* pl  = lse + (size_t)b0 * TLEN;          // lse rows r = 0..7

  // ---- t = 0: alpha2_0 = log2start + 0.8*log2e*em - lsePre ----
  {
    const float ls = logstart[nc];
#pragma unroll
    for (int r = 0; r < 8; ++r) {
      const float e0 = (r < 4) ? pe0[(size_t)r * TSTRIDE] : pe4[(size_t)(r - 4) * TSTRIDE];
      const float l0 = pl[(size_t)r * TLEN];
      const float v  = ls + __builtin_fmaf(EMW_LOG2E, e0, -l0);
      aval[r] = nvalid ? v : -INFINITY;
    }
  }
  tail();

  // software pipeline: preload emissions/lse for t = 1
  float evb[2][8], lvb[2][8];
  pe0 += S_DIM; pe4 += S_DIM; pl += 1;
#pragma unroll
  for (int r = 0; r < 8; ++r) {
    evb[1][r] = (r < 4) ? pe0[(size_t)r * TSTRIDE] : pe4[(size_t)(r - 4) * TSTRIDE];
    lvb[1][r] = pl[(size_t)r * TLEN];
  }

  // ---- serial scan over time ----
#pragma unroll 2
  for (int t = 1; t < TLEN; ++t) {
    const int cur = t & 1, nxt = cur ^ 1;
    // A fragments from staged pExp: M = ln16; K runs {32c+16*hi ..+15}
    v16h afrag[6];
    {
      const _Float16* rowp = pExp + ln16 * PPAD;
#pragma unroll
      for (int c = 0; c < 6; ++c) {
        const v8h* a0 = (const v8h*)(rowp + 32 * c + hi * 16);
        const v8h* a1 = (const v8h*)(rowp + 32 * c + hi * 16 + 8);
        v16h f;
        ((v8h*)&f)[0] = a0[0];
        ((v8h*)&f)[1] = a1[0];
        afrag[c] = f;
      }
    }
    // issue NEXT step's emission/lse loads now (hidden behind WMMA + tail + barriers)
    pe0 += S_DIM; pe4 += S_DIM; pl += 1;
#pragma unroll
    for (int r = 0; r < 8; ++r) {
      evb[nxt][r] = (r < 4) ? pe0[(size_t)r * TSTRIDE] : pe4[(size_t)(r - 4) * TSTRIDE];
      lvb[nxt][r] = pl[(size_t)r * TLEN];
    }
    // speculative prefetch of t+2 cachelines (OOB prefetch is dropped by HW)
    __builtin_prefetch(pe0 + S_DIM, 0, 3);
    __builtin_prefetch(pe4 + S_DIM, 0, 3);
    // s = p @ trans  (f32 accumulate)
    v8f acc = {};
#pragma unroll
    for (int c = 0; c < 6; ++c)
      acc = __builtin_amdgcn_wmma_f32_16x16x32_f16(
          false, afrag[c], false, bfrag[c], (short)0, acc, false, false);
    // alpha2' = log2(s) + 0.8*log2e*em - lsePre  (branchless select)
#pragma unroll
    for (int r = 0; r < 8; ++r) {
      const float v = fast_log2(acc[r]) + __builtin_fmaf(EMW_LOG2E, evb[cur][r], -lvb[cur][r]);
      aval[r] = nvalid ? v : -INFINITY;
    }
    tail();
  }

  // ---- finalize: nll[b] = -ln2 * (log2 sum exp2(alpha2 - m2) + carry2)
  //      carry2 already includes the last m2; mLds holds it for the scaled sum.
  if (tid < 16) {
    const int row = tid;
    const float m = mLds[row];
    float ssum = 0.f;
    for (int k = 0; k < S_DIM; ++k)
      ssum += fast_exp2(alphaLds[row * APAD + k] - m);
    out[wg * 16 + row] = -LN2 * (fast_log2(ssum) + carry[row]);
  }
}

// ---------------------------------------------------------------------------
extern "C" void kernel_launch(void* const* d_in, const int* in_sizes, int n_in,
                              void* d_out, int out_size, void* d_ws, size_t ws_size,
                              hipStream_t stream) {
  const float* em = (const float*)d_in[0];        // [32,4000,170]
  const float* sp = (const float*)d_in[1];        // [170]
  const float* rt = (const float*)d_in[2];        // [170,170]
  float* out = (float*)d_out;                     // [32]

  char* ws = (char*)d_ws;
  _Float16* trans_w  = (_Float16*)(ws + WS_TRANS);
  float*    logstart = (float*)(ws + WS_LOGSTART);
  float*    lse      = (float*)(ws + WS_LSE);

  // zero padded transition matrix (176*192 halves = 16896 dwords)
  zero_u32_kernel<<<(16896 + 255) / 256, 256, 0, stream>>>((uint32_t*)trans_w, 16896);
  trans_kernel<<<S_DIM, 32, 0, stream>>>(rt, trans_w);
  logstart_kernel<<<1, 32, 0, stream>>>(sp, logstart);
  emlse_kernel<<<(BATCH * TLEN) / 8, 256, 0, stream>>>(em, lse);
  scan_kernel<<<2, NTHREADS, 0, stream>>>(em, logstart, trans_w, lse, out);
}
